// WaveNetVAE_61125974556807
// MI455X (gfx1250) — matrix operand order repaired and checked by reading, not compiled, not run on hardware
//
#include <hip/hip_runtime.h>
#include <hip/hip_bf16.h>

typedef __attribute__((ext_vector_type(16))) _Float16 v16h;
typedef __attribute__((ext_vector_type(8)))  _Float16 v8h;
typedef __attribute__((ext_vector_type(8)))  float    v8f;
typedef __attribute__((ext_vector_type(4)))  unsigned v4u;
typedef __attribute__((ext_vector_type(8)))  int      v8i;
typedef __attribute__((ext_vector_type(4)))  int      v4i;

#define C_CH    256
#define TLEN    4096
#define NBATCH  4
#define NTOT    (NBATCH*TLEN)     // 16384 time columns total
#define GUARD   512               // zero guard rows per batch (>= max dilation)
#define RPB     (GUARD + TLEN)    // padded rows per batch
#define TOTROWS (NBATCH*RPB)      // 18432
#define NTILE   64                // time rows per workgroup
#define LDH     264               // halves per LDS tile row (512B data + 16B pad)
#define NBLK    40

// ---------------------------------------------------------------------------
// WMMA fragment loaders per CDNA5 ISA §7.12.2 (wave32).
// A (16x32 f16): lane L -> M=L%16; halves j<8 -> K=j+8*(L/16), j>=8 -> K=16+(j-8)+8*(L/16)
// B (32x16 f16): lane L -> N=L%16; halves i<8 -> K=i+8*(L/16), i>=8 -> K=16+(i-8)+8*(L/16)  (needs K-contiguous rows)
// C/D (16x16 f32): reg r -> M=r+8*(L/16), N=L%16
// h lives time-major [row=t][c], so channel K is contiguous: B loads are two b128.
// ---------------------------------------------------------------------------
static __device__ inline v16h load_a_tile(const _Float16* __restrict__ W, int ldw,
                                          int m0, int k0, int lane) {
  int g = lane >> 4;
  int m = m0 + (lane & 15);
  const _Float16* row = W + m * ldw + k0 + 8 * g;
  union { v16h v; uint4 q[2]; } u;
  u.q[0] = *(const uint4*)(row);
  u.q[1] = *(const uint4*)(row + 16);
  return u.v;
}

static __device__ inline v16h load_b_tile(const _Float16* __restrict__ L, int ldb,
                                          int n0, int k0, int lane) {
  int g = lane >> 4;
  int n = n0 + (lane & 15);
  const _Float16* p = L + n * ldb + k0 + 8 * g;
  union { v16h v; uint4 q[2]; } u;
  u.q[0] = *(const uint4*)(p);
  u.q[1] = *(const uint4*)(p + 16);
  return u.v;
}

static __device__ inline unsigned lds_off_u32(const void* p) {
  return (unsigned)(unsigned long long)(__attribute__((address_space(3))) const char*)p;
}

// ---------------------------------------------------------------------------
// Tensor Data Mover: one 2-D tile load (rows x 512B) global -> LDS, with
// pad_enable inserting 16B after every 512B row (LDS row stride = LDH halves).
// D# packing per ISA §8.3/§8.4.
// ---------------------------------------------------------------------------
static __device__ inline void tdm_load_tile(unsigned lds_addr, const _Float16* gaddr,
                                            int rows) {
  unsigned long long ga = (unsigned long long)(size_t)gaddr;
  v4u g0;
  g0[0] = 1u;                                              // count=1 (user descriptor)
  g0[1] = lds_addr;                                        // lds_addr (bytes)
  g0[2] = (unsigned)(ga & 0xFFFFFFFFu);                    // global_addr[31:0]
  g0[3] = (unsigned)((ga >> 32) & 0x01FFFFFFu) | (2u << 30); // addr[56:32] | type=2
  v8i g1;
  g1[0] = (1 << 16)        // data_size = 2 bytes
        | (1 << 20)        // pad_enable
        | (6 << 22)        // pad_interval: 128 DWORDs (one 512B row)
        | (3 << 25);       // pad_amount: 4 DWORDs (16B)
  g1[1] = (int)(256u << 16);                               // tensor_dim0 = 256 (lo16)
  g1[2] = (int)((TOTROWS & 0xFFFF) << 16);                 // dim0 hi=0 | tensor_dim1 lo
  g1[3] = (int)((TOTROWS >> 16) & 0xFFFF) | (256 << 16);   // tensor_dim1 hi | tile_dim0=256
  g1[4] = rows & 0xFFFF;                                   // tile_dim1 | tile_dim2=0
  g1[5] = 256;                                             // tensor_dim0_stride lo32
  g1[6] = 0;                                               // stride hi | dim1_stride lo
  g1[7] = 0;
  v4i z4 = {0, 0, 0, 0};
#if defined(__clang_major__) && (__clang_major__ >= 23)
  v8i z8 = {0, 0, 0, 0, 0, 0, 0, 0};
  __builtin_amdgcn_tensor_load_to_lds(g0, g1, z4, z4, z8, 0);
#else
  __builtin_amdgcn_tensor_load_to_lds(g0, g1, z4, z4, 0);
#endif
}

// ---------------------------------------------------------------------------
// Weight repack: enc_Wf/Wg [40][256][256][2] f32 -> [40][2][256][256] f16
// ---------------------------------------------------------------------------
__global__ void k_pack_tap(const float* __restrict__ src, _Float16* __restrict__ dst) {
  int idx = blockIdx.x * blockDim.x + threadIdx.x;
  if (idx >= NBLK * 65536) return;
  int bl = idx >> 16, r = idx & 65535;
  dst[bl * 131072 + r]         = (_Float16)src[idx * 2 + 0];
  dst[bl * 131072 + 65536 + r] = (_Float16)src[idx * 2 + 1];
}

__global__ void k_cast_h(const float* __restrict__ src, _Float16* __restrict__ dst, int n) {
  int idx = blockIdx.x * blockDim.x + threadIdx.x;
  if (idx < n) dst[idx] = (_Float16)src[idx];
}

// ---------------------------------------------------------------------------
// Input conv (1 -> 256, k=2, causal) into time-major padded h: [TOTROWS][256] f16.
// One block per time step; guard rows stay zero (memset).
// ---------------------------------------------------------------------------
__global__ void k_in_conv(const float* __restrict__ x, const float* __restrict__ W,
                          const float* __restrict__ bias, _Float16* __restrict__ hT) {
  int n = blockIdx.x;            // 0..NTOT-1
  int c = threadIdx.x;           // 0..255
  int t = n & (TLEN - 1);
  int gr = (n >> 12) * RPB + GUARD + t;
  float v = W[c * 2 + 1] * x[n] + bias[c];
  if (t > 0) v += W[c * 2 + 0] * x[n - 1];
  hT[(size_t)gr * C_CH + c] = (_Float16)v;
}

// ---------------------------------------------------------------------------
// One gated residual block.  256 threads = 8 waves; NTILE=64 time rows,
// all 256 channels.  Wave w owns channels [32w, 32w+32) for f, g and res.
// ---------------------------------------------------------------------------
__global__ __launch_bounds__(256)
void k_gated_block(const _Float16* __restrict__ hsrc, _Float16* __restrict__ hdst,
                   const _Float16* __restrict__ WfH,   // [2][256][256] tap-major f16
                   const _Float16* __restrict__ WgH,
                   const _Float16* __restrict__ WrH,   // [256][256]
                   const float* __restrict__ bf, const float* __restrict__ bg,
                   const float* __restrict__ br,
                   float* __restrict__ outsum,         // [4][256]
                   int dil) {
  extern __shared__ char smem[];
  _Float16* lh0  = (_Float16*)smem;            // shifted h (tap 0), [64][LDH]
  _Float16* lh1  = lh0 + NTILE * LDH;          // h (tap 1)
  _Float16* lout = lh1 + NTILE * LDH;          // gated activations

  const int tid  = threadIdx.x;
  const int lane = tid & 31;
  const int w    = tid >> 5;
  const int gsel = lane >> 4;
  const int nl   = lane & 15;

  const int n0g = blockIdx.x * NTILE;          // global time index of tile
  const int t0  = n0g & (TLEN - 1);
  const int bb  = n0g / TLEN;
  const int gr0 = bb * RPB + GUARD + t0;       // padded row of first tile row

  // ---- TDM: DMA both h tiles into LDS (tap0 row offset = -dil, lands in guard) ----
  if (w == 0) {
    tdm_load_tile(lds_off_u32(lh1), hsrc + (size_t)gr0 * C_CH, NTILE);
  } else if (w == 1) {
    tdm_load_tile(lds_off_u32(lh0), hsrc + (size_t)(gr0 - dil) * C_CH, NTILE);
  }
  __builtin_amdgcn_s_wait_tensorcnt(0);
  __syncthreads();

  // ---- stage 1: f and g pre-activations for this wave's 32 channels ------
  v8f af[2][4], ag[2][4];
  #pragma unroll
  for (int mt = 0; mt < 2; ++mt)
    #pragma unroll
    for (int nt = 0; nt < 4; ++nt) { af[mt][nt] = (v8f){}; ag[mt][nt] = (v8f){}; }

  for (int kt = 0; kt < 8; ++kt) {
    int k0 = kt * 32;
    v16h b0[4], b1[4];
    #pragma unroll
    for (int nt = 0; nt < 4; ++nt) {
      b0[nt] = load_b_tile(lh0, LDH, nt * 16, k0, lane);
      b1[nt] = load_b_tile(lh1, LDH, nt * 16, k0, lane);
    }
    #pragma unroll
    for (int mt = 0; mt < 2; ++mt) {
      int m0 = w * 32 + mt * 16;
      v16h fa0 = load_a_tile(WfH,         256, m0, k0, lane);
      v16h fa1 = load_a_tile(WfH + 65536, 256, m0, k0, lane);
      v16h ga0 = load_a_tile(WgH,         256, m0, k0, lane);
      v16h ga1 = load_a_tile(WgH + 65536, 256, m0, k0, lane);
      #pragma unroll
      for (int nt = 0; nt < 4; ++nt) {
        af[mt][nt] = __builtin_amdgcn_wmma_f32_16x16x32_f16(
            false, fa0, false, b0[nt], (short)0, af[mt][nt], false, false);
        af[mt][nt] = __builtin_amdgcn_wmma_f32_16x16x32_f16(
            false, fa1, false, b1[nt], (short)0, af[mt][nt], false, false);
        ag[mt][nt] = __builtin_amdgcn_wmma_f32_16x16x32_f16(
            false, ga0, false, b0[nt], (short)0, ag[mt][nt], false, false);
        ag[mt][nt] = __builtin_amdgcn_wmma_f32_16x16x32_f16(
            false, ga1, false, b1[nt], (short)0, ag[mt][nt], false, false);
      }
    }
  }

  // ---- activation in registers; packed b128 stores into lout -------------
  #pragma unroll
  for (int mt = 0; mt < 2; ++mt) {
    int c0 = w * 32 + mt * 16 + 8 * gsel;
    const float* bfp = bf + c0;
    const float* bgp = bg + c0;
    #pragma unroll
    for (int nt = 0; nt < 4; ++nt) {
      union { v8h h; uint4 q; } u;
      #pragma unroll
      for (int r = 0; r < 8; ++r) {
        float pf = af[mt][nt][r] + bfp[r];
        float pg = ag[mt][nt][r] + bgp[r];
        u.h[r] = (_Float16)(tanhf(pf) * (1.f / (1.f + __expf(-pg))));
      }
      int row = nt * 16 + nl;
      *(uint4*)(lout + row * LDH + c0) = u.q;
    }
  }
  __syncthreads();

  // ---- skip-path linearization: per-channel column sums of out -----------
  {
    float s = 0.f;
    for (int n = 0; n < NTILE; ++n) s += (float)lout[n * LDH + tid];
    atomicAdd(&outsum[bb * 256 + tid], s);
  }

  // ---- stage 2: residual GEMM h_new = Wr @ out + br + h ------------------
  v8f ar[2][4];
  #pragma unroll
  for (int mt = 0; mt < 2; ++mt)
    #pragma unroll
    for (int nt = 0; nt < 4; ++nt) ar[mt][nt] = (v8f){};

  for (int kt = 0; kt < 8; ++kt) {
    int k0 = kt * 32;
    v16h bt[4];
    #pragma unroll
    for (int nt = 0; nt < 4; ++nt)
      bt[nt] = load_b_tile(lout, LDH, nt * 16, k0, lane);
    #pragma unroll
    for (int mt = 0; mt < 2; ++mt) {
      v16h a = load_a_tile(WrH, 256, w * 32 + mt * 16, k0, lane);
      #pragma unroll
      for (int nt = 0; nt < 4; ++nt)
        ar[mt][nt] = __builtin_amdgcn_wmma_f32_16x16x32_f16(
            false, a, false, bt[nt], (short)0, ar[mt][nt], false, false);
    }
  }

  #pragma unroll
  for (int mt = 0; mt < 2; ++mt) {
    int c0 = w * 32 + mt * 16 + 8 * gsel;
    const float* brp = br + c0;
    #pragma unroll
    for (int nt = 0; nt < 4; ++nt) {
      int row = nt * 16 + nl;
      v8h hres = *(const v8h*)(lh1 + row * LDH + c0);
      union { v8h h; uint4 q; } u;
      #pragma unroll
      for (int r = 0; r < 8; ++r)
        u.h[r] = (_Float16)(ar[mt][nt][r] + brp[r] + (float)hres[r]);
      *(uint4*)(hdst + (size_t)(gr0 + row) * C_CH + c0) = u.q;
    }
  }
}

// ---------------------------------------------------------------------------
// Finalize: pooled = (sum_bl Ws_bl @ outsum_bl)/T + sum_bl bs_bl, latent,
// reparameterize, length-1 decoder (tap-0 always hits causal zero padding,
// so each decoder conv is a single 256x256 GEMV with W[:,:,1]).
// ---------------------------------------------------------------------------
__global__ __launch_bounds__(256)
void k_finalize(const float* __restrict__ outsum,
                const float* __restrict__ Ws, const float* __restrict__ bs,
                const float* __restrict__ fc_mu_W, const float* __restrict__ fc_mu_b,
                const float* __restrict__ fc_lv_W, const float* __restrict__ fc_lv_b,
                const float* __restrict__ eps,
                const float* __restrict__ dec_in_W, const float* __restrict__ dec_in_b,
                const float* __restrict__ dWf, const float* __restrict__ dbf,
                const float* __restrict__ dWg, const float* __restrict__ dbg,
                const float* __restrict__ dWr, const float* __restrict__ dbr,
                const float* __restrict__ outW, const float* __restrict__ outb,
                float* __restrict__ out)   // y(4) | mu(512) | logvar(512)
{
  __shared__ float pooled[4][256];
  __shared__ float zl[4][128];
  __shared__ float h2[4][256];
  __shared__ float go[4][256];
  const int tid = threadIdx.x;

  {
    int c = tid;
    float sbs = 0.f;
    for (int bl = 0; bl < NBLK; ++bl) sbs += bs[bl * 256 + c];
    for (int b = 0; b < 4; ++b) {
      float s = 0.f;
      for (int bl = 0; bl < NBLK; ++bl) {
        const float* wrow = Ws + (size_t)(bl * 256 + c) * 256;
        const float* vv   = outsum + (bl * 4 + b) * 256;
        for (int k = 0; k < 256; ++k) s += wrow[k] * vv[k];
      }
      pooled[b][c] = s * (1.f / TLEN) + sbs;
    }
  }
  __syncthreads();

  for (int it = 0; it < 2; ++it) {
    int idx = it * 256 + tid;
    int b = idx >> 7, l = idx & 127;
    float mu = fc_mu_b[l], lv = fc_lv_b[l];
    for (int c = 0; c < 256; ++c) {
      float p = pooled[b][c];
      mu += fc_mu_W[l * 256 + c] * p;
      lv += fc_lv_W[l * 256 + c] * p;
    }
    out[4 + idx]       = mu;
    out[4 + 512 + idx] = lv;
    zl[b][l] = mu + eps[b * 128 + l] * __expf(0.5f * lv);
  }
  __syncthreads();

  {
    int c = tid;
    for (int b = 0; b < 4; ++b) {
      float s = dec_in_b[c];
      for (int l = 0; l < 128; ++l) s += dec_in_W[c * 128 + l] * zl[b][l];
      h2[b][c] = s;
    }
  }
  __syncthreads();

  for (int i = 0; i < 10; ++i) {
    int c = tid;
    for (int b = 0; b < 4; ++b) {
      float f = dbf[i * 256 + c], g = dbg[i * 256 + c];
      for (int k = 0; k < 256; ++k) {
        float hv = h2[b][k];
        f += dWf[((size_t)(i * 256 + c) * 256 + k) * 2 + 1] * hv;
        g += dWg[((size_t)(i * 256 + c) * 256 + k) * 2 + 1] * hv;
      }
      go[b][c] = tanhf(f) * (1.f / (1.f + __expf(-g)));
    }
    __syncthreads();
    float resv[4];
    for (int b = 0; b < 4; ++b) {
      float s = dbr[i * 256 + c];
      for (int k = 0; k < 256; ++k) s += dWr[(size_t)(i * 256 + c) * 256 + k] * go[b][k];
      resv[b] = s + h2[b][c];
    }
    __syncthreads();
    for (int b = 0; b < 4; ++b) h2[b][c] = resv[b];
    __syncthreads();
  }

  if (tid < 4) {
    float s = outb[0];
    for (int c = 0; c < 256; ++c) s += outW[c] * h2[tid][c];
    out[tid] = s;
  }
}

// ---------------------------------------------------------------------------
extern "C" void kernel_launch(void* const* d_in, const int* in_sizes, int n_in,
                              void* d_out, int out_size, void* d_ws, size_t ws_size,
                              hipStream_t stream) {
  const float* x        = (const float*)d_in[0];
  const float* eps      = (const float*)d_in[1];
  const float* enc_in_W = (const float*)d_in[2];
  const float* enc_in_b = (const float*)d_in[3];
  const float* enc_Wf   = (const float*)d_in[4];
  const float* enc_bf   = (const float*)d_in[5];
  const float* enc_Wg   = (const float*)d_in[6];
  const float* enc_bg   = (const float*)d_in[7];
  const float* enc_Wr   = (const float*)d_in[8];
  const float* enc_br   = (const float*)d_in[9];
  const float* enc_Ws   = (const float*)d_in[10];
  const float* enc_bs   = (const float*)d_in[11];
  const float* fc_mu_W  = (const float*)d_in[12];
  const float* fc_mu_b  = (const float*)d_in[13];
  const float* fc_lv_W  = (const float*)d_in[14];
  const float* fc_lv_b  = (const float*)d_in[15];
  const float* dec_in_W = (const float*)d_in[16];
  const float* dec_in_b = (const float*)d_in[17];
  const float* dec_Wf   = (const float*)d_in[18];
  const float* dec_bf   = (const float*)d_in[19];
  const float* dec_Wg   = (const float*)d_in[20];
  const float* dec_bg   = (const float*)d_in[21];
  const float* dec_Wr   = (const float*)d_in[22];
  const float* dec_br   = (const float*)d_in[23];
  const float* out_W    = (const float*)d_in[24];
  const float* out_b    = (const float*)d_in[25];

  // ---- workspace carving ----
  const size_t hbytes = (size_t)TOTROWS * C_CH * sizeof(_Float16);   // 9.44 MB
  char* ws = (char*)d_ws;
  size_t off = 0;
  _Float16* hA  = (_Float16*)(ws + off); off += hbytes;
  _Float16* hB  = (_Float16*)(ws + off); off += hbytes;
  _Float16* WfH = (_Float16*)(ws + off); off += (size_t)NBLK * 131072 * 2;
  _Float16* WgH = (_Float16*)(ws + off); off += (size_t)NBLK * 131072 * 2;
  _Float16* WrH = (_Float16*)(ws + off); off += (size_t)NBLK * 65536 * 2;
  float* outsum = (float*)(ws + off);    off += (size_t)NBLK * 4 * 256 * 4;

  // ---- zero h buffers (guard bands must be 0) + outsum ----
  hipMemsetAsync(hA, 0, hbytes, stream);
  hipMemsetAsync(hB, 0, hbytes, stream);
  hipMemsetAsync(outsum, 0, (size_t)NBLK * 4 * 256 * 4, stream);

  // ---- weight repack to f16 ----
  {
    int n = NBLK * 65536;
    k_pack_tap<<<(n + 255) / 256, 256, 0, stream>>>(enc_Wf, WfH);
    k_pack_tap<<<(n + 255) / 256, 256, 0, stream>>>(enc_Wg, WgH);
    k_cast_h <<<(n + 255) / 256, 256, 0, stream>>>(enc_Wr, WrH, n);
  }

  // ---- input conv ----
  k_in_conv<<<NTOT, 256, 0, stream>>>(x, enc_in_W, enc_in_b, hA);

  // ---- 40 encoder blocks (serial; L2-resident ping-pong) ----
  const size_t smem = (size_t)3 * NTILE * LDH * sizeof(_Float16);  // ~99 KB
  static const int dils[10] = {1, 2, 4, 8, 16, 32, 64, 128, 256, 512};
  _Float16* hs = hA;
  _Float16* hd = hB;
  for (int bl = 0; bl < NBLK; ++bl) {
    k_gated_block<<<NTOT / NTILE, 256, smem, stream>>>(
        hs, hd,
        WfH + (size_t)bl * 131072, WgH + (size_t)bl * 131072,
        WrH + (size_t)bl * 65536,
        enc_bf + bl * 256, enc_bg + bl * 256, enc_br + bl * 256,
        outsum + bl * 1024, dils[bl % 10]);
    _Float16* t = hs; hs = hd; hd = t;
  }

  // ---- pooled / latent / decoder / output ----
  k_finalize<<<1, 256, 0, stream>>>(outsum, enc_Ws, enc_bs,
                                    fc_mu_W, fc_mu_b, fc_lv_W, fc_lv_b, eps,
                                    dec_in_W, dec_in_b, dec_Wf, dec_bf,
                                    dec_Wg, dec_bg, dec_Wr, dec_br,
                                    out_W, out_b, (float*)d_out);
}
